// FusedMoE_60464549593735
// MI455X (gfx1250) — compile-verified, matrix operand order checked by reading, not run on hardware
//
#include <hip/hip_runtime.h>
#include <hip/hip_bf16.h>
#include <math.h>

#define T_TOK 2048
#define HDIM  1024
#define NEXP  8
#define IDIM  2048
#define TOPK  2
#define CAP   (T_TOK * TOPK + 64)   // padded slot capacity

typedef __attribute__((ext_vector_type(16))) __bf16         v16bf;
typedef __attribute__((ext_vector_type(16))) unsigned short v16u;
typedef __attribute__((ext_vector_type(8)))  float          v8f;
typedef __attribute__((ext_vector_type(2)))  int            v2i;
typedef __attribute__((ext_vector_type(4)))  unsigned int   v4u;
typedef __attribute__((ext_vector_type(8)))  int            v8i;
typedef __attribute__((ext_vector_type(4)))  int            v4i;

typedef __attribute__((address_space(1))) v2i g_v2i;   // global (device) space
typedef __attribute__((address_space(3))) v2i l_v2i;   // LDS space

// ---- feature probes --------------------------------------------------------
#if defined(__has_builtin)
#  if __has_builtin(__builtin_amdgcn_global_load_async_to_lds_b64)
#    define MOE_ASYNC 1
#  endif
#  if __has_builtin(__builtin_amdgcn_cvt_pk_bf16_f32)
#    define MOE_CVTPK 1
#  endif
#  if __has_builtin(__builtin_amdgcn_tensor_load_to_lds)
#    define MOE_TDM 1
#  endif
#endif
#ifndef MOE_ASYNC
#  define MOE_ASYNC 0
#endif
#ifndef MOE_CVTPK
#  define MOE_CVTPK 0
#endif
#ifndef MOE_TDM
#  define MOE_TDM 0
#endif

__device__ __forceinline__ void copy_b64_g2l(unsigned short* lds_dst,
                                             const unsigned short* gsrc) {
#if MOE_ASYNC
  __builtin_amdgcn_global_load_async_to_lds_b64(
      (g_v2i*)gsrc, (l_v2i*)lds_dst, 0, 0);
#else
  *(uint2*)lds_dst = *(const uint2*)gsrc;
#endif
}

__device__ __forceinline__ void wait_async0() {
#if MOE_ASYNC
#  if __has_builtin(__builtin_amdgcn_s_wait_asynccnt)
  __builtin_amdgcn_s_wait_asynccnt(0);
#  else
  asm volatile("s_wait_asynccnt 0x0" ::: "memory");
#  endif
#endif
}

// ---- TDM: one tensor_load_to_lds for a 2D tile (rows x 32 bf16, padded 40) --
// D# pad: after every 16 DWORDs (one 64B row) insert 4 DWORDs -> LDS stride 40.
__device__ __forceinline__ void tdm_load_a_tile(const unsigned short* gsrc,
                                                unsigned short* lds_dst,
                                                unsigned rows,
                                                unsigned row_stride_elems) {
#if MOE_TDM
  unsigned long long ga = (unsigned long long)gsrc;
  unsigned lds_off = (unsigned)(unsigned long long)(void*)lds_dst;
  v4u g0;
  g0[0] = 1u;                                             // count=1 (valid)
  g0[1] = lds_off;                                        // lds_addr
  g0[2] = (unsigned)(ga & 0xffffffffu);                   // global_addr[31:0]
  g0[3] = (unsigned)((ga >> 32) & 0x01ffffffu) | (2u << 30);  // ga[56:32] | type=2
  v8i g1;
  g1[0] = (int)((1u << 16) | (1u << 20) | (3u << 22) | (3u << 25));
          // data_size=2B | pad_enable | pad_interval=16DW | pad_amount=4DW
  g1[1] = (int)((row_stride_elems & 0xffffu) << 16);      // tensor_dim0[15:0]
  g1[2] = (int)((row_stride_elems >> 16) | ((unsigned)(CAP & 0xffff) << 16));
  g1[3] = (int)(((unsigned)CAP >> 16) | (32u << 16));     // tile_dim0 = 32
  g1[4] = (int)rows;                                      // tile_dim1, tile_dim2=0
  g1[5] = (int)row_stride_elems;                          // tensor_dim0_stride lo
  g1[6] = 0;
  g1[7] = 0;
  v4i z4 = {};
#  if __clang_major__ >= 23
  v8i z8 = {};
  __builtin_amdgcn_tensor_load_to_lds(g0, g1, z4, z4, z8, 0);
#  else
  __builtin_amdgcn_tensor_load_to_lds(g0, g1, z4, z4, 0);
#  endif
#else
  (void)gsrc; (void)lds_dst; (void)rows; (void)row_stride_elems;
#endif
}

__device__ __forceinline__ void wait_tensor0() {
#if MOE_TDM
#  if __has_builtin(__builtin_amdgcn_s_wait_tensorcnt)
  __builtin_amdgcn_s_wait_tensorcnt(0);
#  else
  asm volatile("s_wait_tensorcnt 0x0" ::: "memory");
#  endif
#endif
}

// Native hardware bf16 convert (RNE in HW).
__device__ __forceinline__ unsigned short f2bf(float f) {
  return __builtin_bit_cast(unsigned short, (__bf16)f);
}
__device__ __forceinline__ unsigned packbf2(float a, float b) {
#if MOE_CVTPK
  return __builtin_bit_cast(unsigned, __builtin_amdgcn_cvt_pk_bf16_f32(a, b));
#else
  return (unsigned)f2bf(a) | ((unsigned)f2bf(b) << 16);
#endif
}
// Convert 8 floats (two float4) -> 8 packed bf16 in a uint4 (one b128 store).
__device__ __forceinline__ uint4 cvt8(float4 q0, float4 q1) {
  return make_uint4(packbf2(q0.x, q0.y), packbf2(q0.z, q0.w),
                    packbf2(q1.x, q1.y), packbf2(q1.z, q1.w));
}

__device__ __forceinline__ v8f wmma_bf16(v16u a, v16u b, v8f c) {
  return __builtin_amdgcn_wmma_f32_16x16x32_bf16(
      false, __builtin_bit_cast(v16bf, a),
      false, __builtin_bit_cast(v16bf, b),
      (short)0, c, false, false);
}

// Load one 16(row/col) x 32(K) bf16 fragment from LDS, row stride = 40 ushorts.
__device__ __forceinline__ v16u load_frag(const unsigned short* base, int lane) {
  int half = (lane >> 4) & 1;
  int r    = lane & 15;
  const uint4* p = (const uint4*)(base + r * 40 + half * 8);
  union { v16u v; uint4 q[2]; } u;
  u.q[0] = p[0];   // K +0..7
  u.q[1] = p[2];   // K +16..23
  return u.v;
}

// ---------------------------------------------------------------- init
__global__ void moe_init(float* out, int* counts, int* cursor,
                         int* tok_ids, float* tok_w) {
  int i = blockIdx.x * blockDim.x + threadIdx.x;
  if (i < T_TOK * HDIM) out[i] = 0.0f;
  if (i < NEXP) { counts[i] = 0; cursor[i] = 0; }
  if (i < CAP)  { tok_ids[i] = 0; tok_w[i] = 0.0f; }
}

// ---------------------------------------------------------------- router
__global__ void moe_router(const float* __restrict__ logits,
                           int* __restrict__ topk_id, float* __restrict__ topk_w,
                           int* __restrict__ counts) {
  int t = blockIdx.x * blockDim.x + threadIdx.x;
  if (t >= T_TOK) return;
  float l[NEXP];
  float m = -1e30f;
  #pragma unroll
  for (int e = 0; e < NEXP; ++e) { l[e] = logits[t * NEXP + e]; m = fmaxf(m, l[e]); }
  #pragma unroll
  for (int e = 0; e < NEXP; ++e) l[e] = __expf(l[e] - m);
  int i0 = 0; float p0 = l[0];
  #pragma unroll
  for (int e = 1; e < NEXP; ++e) if (l[e] > p0) { p0 = l[e]; i0 = e; }
  int i1 = -1; float p1 = -1.0f;
  #pragma unroll
  for (int e = 0; e < NEXP; ++e) if (e != i0 && l[e] > p1) { p1 = l[e]; i1 = e; }
  float inv = __builtin_amdgcn_rcpf(p0 + p1);
  topk_id[t * 2 + 0] = i0; topk_w[t * 2 + 0] = p0 * inv;
  topk_id[t * 2 + 1] = i1; topk_w[t * 2 + 1] = p1 * inv;
  atomicAdd(&counts[i0], 1);
  atomicAdd(&counts[i1], 1);
}

// ---------------------------------------------------------------- scan (E=8)
__global__ void moe_scan(const int* __restrict__ counts, int* __restrict__ offsets) {
  if (threadIdx.x == 0 && blockIdx.x == 0) {
    int acc = 0;
    for (int e = 0; e < NEXP; ++e) { offsets[e] = acc; acc += counts[e]; }
    offsets[NEXP] = acc;
  }
}

// ---------------------------------------------------------------- scatter
__global__ void moe_scatter(const int* __restrict__ topk_id, const float* __restrict__ topk_w,
                            const int* __restrict__ offsets, int* __restrict__ cursor,
                            int* __restrict__ tok_ids, float* __restrict__ tok_w) {
  int t = blockIdx.x * blockDim.x + threadIdx.x;
  if (t >= T_TOK) return;
  #pragma unroll
  for (int k = 0; k < TOPK; ++k) {
    int e    = topk_id[t * 2 + k];
    int pos  = atomicAdd(&cursor[e], 1);
    int slot = offsets[e] + pos;
    tok_ids[slot] = t;
    tok_w[slot]   = topk_w[t * 2 + k];
  }
}

// ---------------------------------------------------------------- hs fp32 -> bf16
__global__ void moe_cvt_hs(const float* __restrict__ hs, unsigned short* __restrict__ hb) {
  size_t i = ((size_t)blockIdx.x * blockDim.x + threadIdx.x) * 8;
  float4 v0 = *(const float4*)(hs + i);
  float4 v1 = *(const float4*)(hs + i + 4);
  *(uint4*)(hb + i) = cvt8(v0, v1);
}

// ---------------------------------------------------------------- GEMM1
// act = silu(x@Wg^T) * (x@Wu^T); grid = (E, T/32, I/128), 256 threads (8 waves).
#define G1_LA (32 * 40)
#define G1_LB (128 * 40)
__global__ __launch_bounds__(256) void moe_gemm1(
    const unsigned short* __restrict__ hs_bf, const float* __restrict__ w13,
    const int* __restrict__ counts, const int* __restrict__ offsets,
    const int* __restrict__ tok_ids, unsigned short* __restrict__ act) {
  int e      = blockIdx.x;
  int cnt    = counts[e];
  int m_base = blockIdx.y * 32;
  if (m_base >= cnt) return;                 // block-uniform
  int off_e  = offsets[e];
  int n0     = blockIdx.z * 128;

  __shared__ unsigned short As[2 * G1_LA];
  __shared__ unsigned short Bg[2 * G1_LB];
  __shared__ unsigned short Bu[2 * G1_LB];

  int tid = threadIdx.x, lane = tid & 31, wave = tid >> 5;
  int m_sub = wave & 1, n_quad = wave >> 1;

  int a_row = tid >> 3, a_kg = tid & 7;
  int tok   = tok_ids[off_e + m_base + a_row];
  const unsigned short* a_src = hs_bf + (size_t)tok * HDIM + a_kg * 4;
  unsigned short* a_dst0 = As + a_row * 40 + a_kg * 4;

  int b_row = tid >> 1, b_k16 = (tid & 1) * 16;
  const float* g_src = w13 + ((size_t)e * 2 * IDIM + (size_t)(n0 + b_row)) * HDIM + b_k16;
  const float* u_src = w13 + ((size_t)e * 2 * IDIM + (size_t)(IDIM + n0 + b_row)) * HDIM + b_k16;
  unsigned short* bg_dst0 = Bg + b_row * 40 + b_k16;
  unsigned short* bu_dst0 = Bu + b_row * 40 + b_k16;

  v8f accg[2] = {};
  v8f accu[2] = {};
  float4 gq[4], uq[4];

  // ---- prologue: stage tile 0 into buffer 0
  copy_b64_g2l(a_dst0, a_src);
  #pragma unroll
  for (int j = 0; j < 4; ++j) {
    gq[j] = *(const float4*)(g_src + j * 4);
    uq[j] = *(const float4*)(u_src + j * 4);
  }
  *(uint4*)(bg_dst0 + 0) = cvt8(gq[0], gq[1]);
  *(uint4*)(bg_dst0 + 8) = cvt8(gq[2], gq[3]);
  *(uint4*)(bu_dst0 + 0) = cvt8(uq[0], uq[1]);
  *(uint4*)(bu_dst0 + 8) = cvt8(uq[2], uq[3]);
  wait_async0();
  __syncthreads();

  const int NSTEP = HDIM / 32;
  #pragma unroll 2
  for (int i = 0; i < NSTEP; ++i) {
    int cur = i & 1, nxt = cur ^ 1;
    int kn  = (i + 1) * 32;
    bool has_next = (i + 1 < NSTEP);

    if (has_next) {
      copy_b64_g2l(a_dst0 + nxt * G1_LA, a_src + kn);
      #pragma unroll
      for (int j = 0; j < 4; ++j) {
        gq[j] = *(const float4*)(g_src + kn + j * 4);
        uq[j] = *(const float4*)(u_src + kn + j * 4);
      }
      __builtin_prefetch(g_src + kn + 32, 0, 3);
      __builtin_prefetch(u_src + kn + 32, 0, 3);
    }

    v16u af  = load_frag(As + cur * G1_LA + m_sub * 16 * 40, lane);
    v16u bg0 = load_frag(Bg + cur * G1_LB + (n_quad * 32 + 0)  * 40, lane);
    v16u bu0 = load_frag(Bu + cur * G1_LB + (n_quad * 32 + 0)  * 40, lane);
    v16u bg1 = load_frag(Bg + cur * G1_LB + (n_quad * 32 + 16) * 40, lane);
    v16u bu1 = load_frag(Bu + cur * G1_LB + (n_quad * 32 + 16) * 40, lane);
    accg[0] = wmma_bf16(af, bg0, accg[0]);
    accu[0] = wmma_bf16(af, bu0, accu[0]);
    accg[1] = wmma_bf16(af, bg1, accg[1]);
    accu[1] = wmma_bf16(af, bu1, accu[1]);

    if (has_next) {
      unsigned short* bgd = bg_dst0 + nxt * G1_LB;
      unsigned short* bud = bu_dst0 + nxt * G1_LB;
      *(uint4*)(bgd + 0) = cvt8(gq[0], gq[1]);
      *(uint4*)(bgd + 8) = cvt8(gq[2], gq[3]);
      *(uint4*)(bud + 0) = cvt8(uq[0], uq[1]);
      *(uint4*)(bud + 8) = cvt8(uq[2], uq[3]);
      wait_async0();
    }
    __syncthreads();
  }

  // Fused SiLU(gate)*up epilogue
  int half = lane >> 4, nloc = lane & 15;
  #pragma unroll
  for (int c = 0; c < 2; ++c) {
    #pragma unroll
    for (int r = 0; r < 8; ++r) {
      int row = m_sub * 16 + r + 8 * half;
      if (m_base + row < cnt) {
        float g = accg[c][r], u = accu[c][r];
        float a = g * u * __builtin_amdgcn_rcpf(1.0f + __expf(-g));
        act[(size_t)(off_e + m_base + row) * IDIM +
            (n0 + n_quad * 32 + c * 16 + nloc)] = f2bf(a);
      }
    }
  }
}

// ---------------------------------------------------------------- GEMM2
// out[t,h] += w * (act @ W2^T); grid = (E, T/32, H/256), 256 threads (8 waves).
// A tile staged by the Tensor Data Mover (one tensor_load_to_lds per K-step).
#define G2_LA (32 * 40)
#define G2_LB (256 * 40)
__global__ __launch_bounds__(256) void moe_gemm2(
    const unsigned short* __restrict__ act, const float* __restrict__ w2,
    const int* __restrict__ counts, const int* __restrict__ offsets,
    const int* __restrict__ tok_ids, const float* __restrict__ tok_w,
    float* __restrict__ out) {
  int e      = blockIdx.x;
  int cnt    = counts[e];
  int m_base = blockIdx.y * 32;
  if (m_base >= cnt) return;
  int off_e  = offsets[e];
  int h0     = blockIdx.z * 256;

  __shared__ unsigned short As[2 * G2_LA];
  __shared__ unsigned short Bs[2 * G2_LB];

  int tid = threadIdx.x, lane = tid & 31, wave = tid >> 5;
  int m_sub = wave & 1, n_quad = wave >> 1;   // wave covers 16 rows x 64 cols

  // A tile source: 32 contiguous act rows, row stride IDIM (2D TDM tile)
  const unsigned short* a_tile = act + (size_t)(off_e + m_base) * IDIM;
#if !MOE_TDM
  int a_row = tid >> 3, a_kg = tid & 7;
  const unsigned short* a_src = a_tile + (size_t)a_row * IDIM + a_kg * 4;
  unsigned short* a_dst0 = As + a_row * 40 + a_kg * 4;
#endif

  // B staging: thread -> one h row (32 K floats = 8x float4)
  const float* b_src = w2 + ((size_t)e * HDIM + (size_t)(h0 + tid)) * IDIM;
  unsigned short* b_dst0 = Bs + tid * 40;

  v8f acc[4] = {};
  float4 bq[8];

  // ---- prologue
#if MOE_TDM
  if (wave == 0) tdm_load_a_tile(a_tile, As, 32, IDIM);
#else
  copy_b64_g2l(a_dst0, a_src);
#endif
  #pragma unroll
  for (int j = 0; j < 8; ++j) bq[j] = *(const float4*)(b_src + j * 4);
  #pragma unroll
  for (int j = 0; j < 4; ++j)
    *(uint4*)(b_dst0 + j * 8) = cvt8(bq[2 * j], bq[2 * j + 1]);
#if MOE_TDM
  if (wave == 0) wait_tensor0();
#else
  wait_async0();
#endif
  __syncthreads();

  const int NSTEP = IDIM / 32;
  #pragma unroll 2
  for (int i = 0; i < NSTEP; ++i) {
    int cur = i & 1, nxt = cur ^ 1;
    int kn  = (i + 1) * 32;
    bool has_next = (i + 1 < NSTEP);

    if (has_next) {
#if MOE_TDM
      if (wave == 0) tdm_load_a_tile(a_tile + kn, As + nxt * G2_LA, 32, IDIM);
#else
      copy_b64_g2l(a_dst0 + nxt * G2_LA, a_src + kn);
#endif
      #pragma unroll
      for (int j = 0; j < 8; ++j) bq[j] = *(const float4*)(b_src + kn + j * 4);
      __builtin_prefetch(b_src + kn + 32, 0, 3);
    }

    v16u af = load_frag(As + cur * G2_LA + m_sub * 16 * 40, lane);
    v16u b0 = load_frag(Bs + cur * G2_LB + (n_quad * 64 + 0)  * 40, lane);
    v16u b1 = load_frag(Bs + cur * G2_LB + (n_quad * 64 + 16) * 40, lane);
    v16u b2 = load_frag(Bs + cur * G2_LB + (n_quad * 64 + 32) * 40, lane);
    v16u b3 = load_frag(Bs + cur * G2_LB + (n_quad * 64 + 48) * 40, lane);
    acc[0] = wmma_bf16(af, b0, acc[0]);
    acc[1] = wmma_bf16(af, b1, acc[1]);
    acc[2] = wmma_bf16(af, b2, acc[2]);
    acc[3] = wmma_bf16(af, b3, acc[3]);

    if (has_next) {
      unsigned short* bd = b_dst0 + nxt * G2_LB;
      #pragma unroll
      for (int j = 0; j < 4; ++j)
        *(uint4*)(bd + j * 8) = cvt8(bq[2 * j], bq[2 * j + 1]);
#if MOE_TDM
      if (wave == 0) wait_tensor0();
#else
      wait_async0();
#endif
    }
    __syncthreads();
  }

  int half = lane >> 4, nloc = lane & 15;
  #pragma unroll
  for (int c = 0; c < 4; ++c) {
    #pragma unroll
    for (int r = 0; r < 8; ++r) {
      int row = m_sub * 16 + r + 8 * half;
      int mg  = m_base + row;
      if (mg < cnt) {
        int slot = off_e + mg;
        atomicAdd(out + (size_t)tok_ids[slot] * HDIM +
                      (h0 + n_quad * 64 + c * 16 + nloc),
                  tok_w[slot] * acc[c][r]);
      }
    }
  }
}

// ---------------------------------------------------------------- launch
extern "C" void kernel_launch(void* const* d_in, const int* in_sizes, int n_in,
                              void* d_out, int out_size, void* d_ws, size_t ws_size,
                              hipStream_t stream) {
  (void)in_sizes; (void)n_in; (void)out_size; (void)ws_size;
  const float* hs     = (const float*)d_in[0];
  const float* logits = (const float*)d_in[1];
  const float* w13    = (const float*)d_in[2];
  const float* w2     = (const float*)d_in[3];
  float* out = (float*)d_out;

  int*   counts  = (int*)d_ws;                              // 16 ints
  int*   cursor  = counts + 16;                             // 16
  int*   offsets = cursor + 16;                             // 16 (E+1 used)
  int*   topk_id = offsets + 16;                            // T*2
  float* topk_w  = (float*)(topk_id + T_TOK * 2);           // T*2
  int*   tok_ids = (int*)(topk_w + T_TOK * 2);              // CAP
  float* tok_wc  = (float*)(tok_ids + CAP);                 // CAP
  unsigned short* hs_bf = (unsigned short*)(tok_wc + CAP);  // T*H bf16
  unsigned short* act   = hs_bf + (size_t)T_TOK * HDIM;     // CAP*I bf16

  moe_init   <<<(T_TOK * HDIM + 255) / 256, 256, 0, stream>>>(out, counts, cursor, tok_ids, tok_wc);
  moe_router <<<(T_TOK + 255) / 256, 256, 0, stream>>>(logits, topk_id, topk_w, counts);
  moe_scan   <<<1, 1, 0, stream>>>(counts, offsets);
  moe_scatter<<<(T_TOK + 255) / 256, 256, 0, stream>>>(topk_id, topk_w, offsets, cursor, tok_ids, tok_wc);
  moe_cvt_hs <<<(T_TOK * HDIM / 8 + 255) / 256, 256, 0, stream>>>(hs, hs_bf);
  moe_gemm1  <<<dim3(NEXP, T_TOK / 32, IDIM / 128), 256, 0, stream>>>(hs_bf, w13, counts, offsets, tok_ids, act);
  moe_gemm2  <<<dim3(NEXP, T_TOK / 32, HDIM / 256), 256, 0, stream>>>(act, w2, counts, offsets, tok_ids, tok_wc, out);
}
